// BispectrumPool3d_66932770341491
// MI455X (gfx1250) — matrix operand order repaired and checked by reading, not compiled, not run on hardware
//
#include <hip/hip_runtime.h>

typedef _Float16 v16h __attribute__((ext_vector_type(16)));
typedef float    v8f  __attribute__((ext_vector_type(8)));
typedef float    v4f  __attribute__((ext_vector_type(4)));
typedef float    v2f  __attribute__((ext_vector_type(2)));

#define GDIM   24
#define SPA    13824          // 24*24*24 spatial sites per (b,c,g)
#define FEAT   72             // G * N_SEL
#define KTOT   2304           // C * FEAT
#define HEADS  64
#define KSTEPS 72             // KTOT / 32
#define TILES_PER_B 864       // SPA / 16
#define LN2    0.69314718055994530942f

// ---- compile-time octahedral (S4) Cayley table, lexicographic perm order ----
struct Tables { int cay[24][24]; int cgs[24][3]; };

constexpr Tables make_tables() {
  Tables t{};
  int perms[24][4] = {};
  int n = 0;
  for (int a = 0; a < 4; ++a)
    for (int b = 0; b < 4; ++b) { if (b == a) continue;
      for (int c = 0; c < 4; ++c) { if (c == a || c == b) continue;
        for (int d = 0; d < 4; ++d) { if (d == a || d == b || d == c) continue;
          perms[n][0] = a; perms[n][1] = b; perms[n][2] = c; perms[n][3] = d; ++n; } } }
  for (int p = 0; p < 24; ++p)
    for (int q = 0; q < 24; ++q) {
      int comp[4]; for (int k = 0; k < 4; ++k) comp[k] = perms[p][perms[q][k]];
      for (int r = 0; r < 24; ++r) {
        bool eq = true;
        for (int k = 0; k < 4; ++k) if (perms[r][k] != comp[k]) { eq = false; break; }
        if (eq) { t.cay[p][q] = r; break; }
      }
    }
  const int selp[3][4] = {{0,1,2,3},{1,0,2,3},{1,2,3,0}};
  int sel[3] = {};
  for (int s = 0; s < 3; ++s)
    for (int r = 0; r < 24; ++r) {
      bool eq = true;
      for (int k = 0; k < 4; ++k) if (perms[r][k] != selp[s][k]) { eq = false; break; }
      if (eq) { sel[s] = r; break; }
    }
  for (int g = 0; g < 24; ++g)
    for (int s = 0; s < 3; ++s) t.cgs[g][s] = t.cay[g][sel[s]];
  return t;
}
constexpr Tables TBL = make_tables();

// ---- forced VOP3P packed-f32 ops (2 FMA/lane/clk path) ----
__device__ __forceinline__ v2f pk_mul(v2f a, v2f b) {
  v2f d;
  asm("v_pk_mul_f32 %0, %1, %2" : "=v"(d) : "v"(a), "v"(b));
  return d;
}
__device__ __forceinline__ void pk_fma_acc(v2f& acc, v2f a, v2f b) {
  asm("v_pk_fma_f32 %0, %1, %2, %0" : "+v"(acc) : "v"(a), "v"(b));
}

// ---- pre-pass: pack ln2*W[64][2304] (f32) into B-fragment order (f16) ----
// B fragment (32x16 f16, ISA 7.12.2): lane = N + 16*((K/8)&1), halfidx = K%8 + 8*(K/16).
// Layout: Wb[kt][ht][lane][j], flat half index = ((kt*4+ht)*32+lane)*16 + j.
// ln2 factor compensates features stored as log2(1+beta) instead of ln(1+beta).
__global__ __launch_bounds__(256) void pack_wb_kernel(const float* __restrict__ W,
                                                      _Float16* __restrict__ Wb) {
  int id = blockIdx.x * 256 + threadIdx.x;           // 0 .. 147455
  int j    = id & 15;
  int lane = (id >> 4) & 31;
  int ht   = (id >> 9) & 3;
  int kt   = id >> 11;
  int o = ht * 16 + (lane & 15);
  int k = kt * 32 + ((lane >> 4) << 3) + (j & 7) + ((j >> 3) << 4);
  Wb[id] = (_Float16)(LN2 * W[o * KTOT + k]);
}

// ---- fused bispectrum + log/relu + WMMA GEMM kernel ----
__global__ __launch_bounds__(128) void bispec_gemm_kernel(
    const float* __restrict__ x, const float* __restrict__ bias,
    const _Float16* __restrict__ Wb, float* __restrict__ out) {
  // A-fragments for 16 positions x 2304 K, already in WMMA lane order.
  __shared__ v16h Afrag[KSTEPS][32];                 // 73,728 bytes

  const int tid  = threadIdx.x;
  const int blk  = blockIdx.x;
  const int bb   = blk / TILES_PER_B;
  const int base = (blk - bb * TILES_PER_B) * 16;    // dhw tile start

  // ------- stage 1: packed-f32 bispectrum (2 positions/thread) -> LDS -------
  #pragma unroll 1
  for (int t = 0; t < 2; ++t) {
    const int pt = t * 128 + tid;                    // packed task 0..255
    const int c  = pt >> 3;                          // channel 0..31
    const int pp = (pt & 7) << 1;                    // position pair {pp, pp+1}
    const float* xp = x + ((size_t)(bb * 32 + c) * GDIM) * SPA + base + pp;

    v2f xf[GDIM];                                    // 2 sites in packed regs
    #pragma unroll
    for (int g = 0; g < GDIM; ++g) xf[g] = *(const v2f*)(xp + (size_t)g * SPA);

    const int k0 = c * FEAT;
    #pragma unroll
    for (int s = 0; s < 3; ++s) {
      v2f acc[GDIM];
      #pragma unroll
      for (int h = 0; h < GDIM; ++h) acc[h] = (v2f)0.0f;
      #pragma unroll
      for (int g = 0; g < GDIM; ++g) {
        const v2f p = pk_mul(xf[g], xf[TBL.cgs[g][s]]);   // v_pk_mul_f32
        #pragma unroll
        for (int h = 0; h < GDIM; ++h)
          pk_fma_acc(acc[h], p, xf[TBL.cay[g][h]]);       // v_pk_fma_f32
      }
      #pragma unroll
      for (int h = 0; h < GDIM; ++h) {
        // relu(sign(b)*log1p(|b|)) == max(log(1+b), 0); store log2 form,
        // ln2 folded into Wb. 1+b<=0 -> v_log gives NaN/-inf -> v_max_num -> 0.
        const float fa = fmaxf(__builtin_amdgcn_logf(acc[h].x + 1.0f), 0.0f);
        const float fb = fmaxf(__builtin_amdgcn_logf(acc[h].y + 1.0f), 0.0f);
        const int k  = k0 + h * 3 + s;
        const int kt = k >> 5;
        const int kk = k & 31;
        // A fragment (16x32 f16): lane = M + 16*((K/8)&1), half = K%8 + 8*(K/16)
        const int lhalf = (((kk >> 3) & 1) << 4);
        const int half  = (kk & 7) + ((kk >> 4) << 3);
        ((_Float16*)&Afrag[kt][pp + lhalf])[half]     = (_Float16)fa;
        ((_Float16*)&Afrag[kt][pp + 1 + lhalf])[half] = (_Float16)fb;
      }
    }
  }
  __syncthreads();

  // ---------- stage 2: 16x64 GEMM tile via v_wmma_f32_16x16x32_f16 ----------
  const int lane = tid & 31;
  const int ht   = tid >> 5;                         // head tile 0..3 (wave id)
  v8f d0 = {}, d1 = {};
  #pragma unroll 4
  for (int kt = 0; kt < KSTEPS; kt += 2) {
    v16h a0 = Afrag[kt][lane];
    v16h b0 = ((const v16h*)Wb)[(kt * 4 + ht) * 32 + lane];
    d0 = __builtin_amdgcn_wmma_f32_16x16x32_f16(false, a0, false, b0,
                                                (short)0, d0, false, false);
    v16h a1 = Afrag[kt + 1][lane];
    v16h b1 = ((const v16h*)Wb)[((kt + 1) * 4 + ht) * 32 + lane];
    d1 = __builtin_amdgcn_wmma_f32_16x16x32_f16(false, a1, false, b1,
                                                (short)0, d1, false, false);
  }
  v8f d = d0 + d1;

  // D fragment: lane L, vgpr r -> M = r + 8*(L/16), N = L%16 (contiguous in dhw)
  const int nn  = lane & 15;
  const int lhi = lane >> 4;
  const int o   = ht * 16 + nn;
  const float bv = bias[o];
  float* op = out + ((size_t)(bb * HEADS + o)) * SPA + base + lhi * 8;
  v4f lo = { d[0] + bv, d[1] + bv, d[2] + bv, d[3] + bv };
  v4f hi = { d[4] + bv, d[5] + bv, d[6] + bv, d[7] + bv };
  *(v4f*)(op)     = lo;
  *(v4f*)(op + 4) = hi;
}

extern "C" void kernel_launch(void* const* d_in, const int* in_sizes, int n_in,
                              void* d_out, int out_size, void* d_ws, size_t ws_size,
                              hipStream_t stream) {
  const float* x    = (const float*)d_in[0];   // (2,32,24,24,24,24) f32
  const float* W    = (const float*)d_in[1];   // (64,2304) f32
  const float* bias = (const float*)d_in[2];   // (64,) f32
  // d_in[3]=cayley, d_in[4]=sel_idx: deterministic, baked in at compile time.
  float* out = (float*)d_out;                  // (2,64,24,24,24) f32

  if (ws_size < (size_t)KSTEPS * 4 * 32 * 16 * sizeof(_Float16)) return;
  _Float16* Wb = (_Float16*)d_ws;

  pack_wb_kernel<<<576, 256, 0, stream>>>(W, Wb);
  bispec_gemm_kernel<<<2 * TILES_PER_B, 128, 0, stream>>>(x, bias, Wb, out);
}
